// ScaledDotProductAttention_68513318306333
// MI455X (gfx1250) — matrix operand order repaired and checked by reading, not compiled, not run on hardware
//
#include <hip/hip_runtime.h>

typedef __attribute__((ext_vector_type(16))) _Float16 v16h;
typedef __attribute__((ext_vector_type(8)))  float    v8f;

namespace {
constexpr int HEADS = 16;
constexpr int SEQ   = 2048;
constexpr int DIM   = 64;
constexpr int KT    = 64;            // keys per mainloop iteration
constexpr int MT    = 2;             // 16-row query tiles per wave
constexpr int WAVES = 8;             // 256 threads = 8 wave32
constexpr int QROWS = MT * 16;       // 32 query rows per wave
constexpr int QBLK  = WAVES * QROWS; // 256 query rows per block
constexpr float SCALE  = 0.125f;     // 1/sqrt(64)
constexpr float NEGBIG = -1.0e9f;
}

__device__ __forceinline__ v8f wmma16x16x32(v16h a, v16h b, v8f c) {
  // D = A(16x32 f16) * B(32x16 f16) + C(16x16 f32)
  return __builtin_amdgcn_wmma_f32_16x16x32_f16(false, a, false, b,
                                                (short)0, c, false, false);
}

__global__ __launch_bounds__(256, 2)
void flash_attn_f16wmma(const float* __restrict__ Qg,
                        const float* __restrict__ Kg,
                        const float* __restrict__ Vg,
                        const int*   __restrict__ Mg,
                        float* __restrict__ Og)
{
  __shared__ _Float16 Klds[KT * DIM];          // [key][d]  row-major, 8 KB
  __shared__ _Float16 VldsT[DIM * KT];         // [d][key]  transposed, 8 KB
  __shared__ _Float16 Plds[WAVES][QROWS * KT]; // per-wave P scratch, 32 KB

  const int tid  = threadIdx.x;
  const int lane = tid & 31;
  const int wave = tid >> 5;
  const int col  = lane & 15;     // N column / M row index inside 16-lane half
  const int hi   = lane >> 4;     // which half of the wave
  const int a0   = hi ? 16 : 0;   // A-layout K offsets (contiguous: k = 16*hi + e)
  const int a1   = hi ? 24 : 8;

  const int bh    = blockIdx.y;           // b*HEADS + h
  const int bIdx  = bh / HEADS;
  const int qbase = blockIdx.x * QBLK + wave * QROWS;

  const float* Qp = Qg + (size_t)bh * SEQ * DIM;
  const float* Kp = Kg + (size_t)bh * SEQ * DIM;
  const float* Vp = Vg + (size_t)bh * SEQ * DIM;
  const int*   Mp = Mg + (size_t)bIdx * SEQ * SEQ;   // mask [q][k]
  float*       Op = Og + (size_t)bh * SEQ * DIM;

  // ---- Q A-fragments: 2 M-tiles x 2 k-chunks, fixed for whole kernel ----
  v16h qa[MT][2];
  #pragma unroll
  for (int t = 0; t < MT; ++t) {
    #pragma unroll
    for (int c = 0; c < 2; ++c) {
      const float* qr = Qp + (size_t)(qbase + t * 16 + col) * DIM + c * 32;
      #pragma unroll
      for (int e = 0; e < 8; ++e) qa[t][c][e]     = (_Float16)qr[a0 + e];
      #pragma unroll
      for (int e = 0; e < 8; ++e) qa[t][c][8 + e] = (_Float16)qr[a1 + e];
    }
  }

  // Online-softmax state; C/D layout: lane slot r holds row (tile*16 + hi*8 + r)
  float mrow[MT][8], lrow[MT][8];
  v8f acc[MT][4];
  #pragma unroll
  for (int t = 0; t < MT; ++t) {
    #pragma unroll
    for (int r = 0; r < 8; ++r) { mrow[t][r] = -3.0e38f; lrow[t][r] = 0.0f; }
    #pragma unroll
    for (int c = 0; c < 4; ++c) acc[t][c] = (v8f){0,0,0,0,0,0,0,0};
  }

  for (int kb = 0; kb < SEQ; kb += KT) {
    // ---- Stage K [key][d] and V transposed [d][key] (f32 -> f16) ----
    {
      const int key = tid >> 2;          // 0..63
      const int d0  = (tid & 3) * 16;    // 0,16,32,48
      const float* ks = Kp + (size_t)(kb + key) * DIM + d0;
      const float* vs = Vp + (size_t)(kb + key) * DIM + d0;
      #pragma unroll
      for (int e = 0; e < 16; ++e) {
        Klds[key * DIM + d0 + e]   = (_Float16)ks[e];
        VldsT[(d0 + e) * KT + key] = (_Float16)vs[e];
      }
      if (kb + KT < SEQ) {  // global_prefetch_b8 for the next tile
        __builtin_prefetch(Kp + (size_t)(kb + KT + key) * DIM + d0, 0, 3);
        __builtin_prefetch(Vp + (size_t)(kb + KT + key) * DIM + d0, 0, 3);
      }
    }
    __syncthreads();

    v16h pa[MT][2];   // P A-fragments for both M tiles, both 32-key chunks

    #pragma unroll
    for (int t = 0; t < MT; ++t) {
      // ---- Scores: S(16x64) = Q(16x64) * K^T(64x64), four 16x16 N-tiles ----
      v8f sc[4];
      #pragma unroll
      for (int j = 0; j < 4; ++j) {
        v8f c = (v8f){0,0,0,0,0,0,0,0};
        #pragma unroll
        for (int cc = 0; cc < 2; ++cc) {
          // B(32x16): n = 16*j + col, contraction d = cc*32 + hi*16 + e
          v16h kbf;
          const _Float16* src = &Klds[(16 * j + col) * DIM + cc * 32 + hi * 16];
          #pragma unroll
          for (int e = 0; e < 16; ++e) kbf[e] = src[e];
          c = wmma16x16x32(qa[t][cc], kbf, c);
        }
        sc[j] = c;
      }

      // ---- Mask + scale + online softmax ----
      float p[4][8];
      #pragma unroll
      for (int r = 0; r < 8; ++r) {
        const int q = qbase + t * 16 + hi * 8 + r;
        const int* mr = Mp + (size_t)q * SEQ + kb;
        float sv[4];
        float mx = -3.0e38f;
        #pragma unroll
        for (int j = 0; j < 4; ++j) {
          float s = sc[j][r] * SCALE;
          s = mr[j * 16 + col] ? s + NEGBIG : s;   // cmp + cndmask, no cvt
          sv[j] = s;
          mx = fmaxf(mx, s);
        }
        mx = fmaxf(mx, __shfl_xor(mx, 1, 32));
        mx = fmaxf(mx, __shfl_xor(mx, 2, 32));
        mx = fmaxf(mx, __shfl_xor(mx, 4, 32));
        mx = fmaxf(mx, __shfl_xor(mx, 8, 32));
        const float mnew  = fmaxf(mrow[t][r], mx);
        const float alpha = __expf(mrow[t][r] - mnew);
        float ps = 0.0f;
        #pragma unroll
        for (int j = 0; j < 4; ++j) {
          const float pe = __expf(sv[j] - mnew);
          p[j][r] = pe;
          ps += pe;
        }
        ps += __shfl_xor(ps, 1, 32);
        ps += __shfl_xor(ps, 2, 32);
        ps += __shfl_xor(ps, 4, 32);
        ps += __shfl_xor(ps, 8, 32);
        lrow[t][r] = lrow[t][r] * alpha + ps;
        mrow[t][r] = mnew;
        #pragma unroll
        for (int c = 0; c < 4; ++c) acc[t][c][r] *= alpha;
      }

      // ---- Relayout P (C/D layout -> A layout) via per-wave LDS scratch ----
      _Float16* pl = &Plds[wave][(t * 16) * KT];
      #pragma unroll
      for (int r = 0; r < 8; ++r) {
        const int m = hi * 8 + r;
        #pragma unroll
        for (int j = 0; j < 4; ++j)
          pl[m * KT + j * 16 + col] = (_Float16)p[j][r];
      }
      // Same-wave DS ops are in order: read back as A fragments.
      #pragma unroll
      for (int kc = 0; kc < 2; ++kc) {
        const _Float16* pr = &pl[col * KT + kc * 32];
        #pragma unroll
        for (int e = 0; e < 8; ++e) pa[t][kc][e]     = pr[a0 + e];
        #pragma unroll
        for (int e = 0; e < 8; ++e) pa[t][kc][8 + e] = pr[a1 + e];
      }
    }

    // ---- O += P(32x64) * V(64x64); V B-fragments reused across M tiles ----
    #pragma unroll
    for (int c = 0; c < 4; ++c) {
      #pragma unroll
      for (int kc = 0; kc < 2; ++kc) {
        // B(32x16): n = c*16 + col (dim), k = kc*32 + hi*16 + e (key)
        v16h vb;
        const _Float16* src = &VldsT[(c * 16 + col) * KT + kc * 32 + hi * 16];
        #pragma unroll
        for (int e = 0; e < 16; ++e) vb[e] = src[e];
        #pragma unroll
        for (int t = 0; t < MT; ++t)
          acc[t][c] = wmma16x16x32(pa[t][kc], vb, acc[t][c]);
      }
    }
    __syncthreads();  // protect K/V LDS tiles before next staging
  }

  // ---- Epilogue: normalize and store ----
  #pragma unroll
  for (int t = 0; t < MT; ++t) {
    #pragma unroll
    for (int r = 0; r < 8; ++r) {
      const int q = qbase + t * 16 + hi * 8 + r;
      const float inv = 1.0f / lrow[t][r];
      #pragma unroll
      for (int c = 0; c < 4; ++c) {
        Op[(size_t)q * DIM + c * 16 + col] = acc[t][c][r] * inv;
      }
    }
  }
}

extern "C" void kernel_launch(void* const* d_in, const int* in_sizes, int n_in,
                              void* d_out, int out_size, void* d_ws, size_t ws_size,
                              hipStream_t stream) {
  (void)in_sizes; (void)n_in; (void)out_size; (void)d_ws; (void)ws_size;
  const float* q = (const float*)d_in[0];
  const float* k = (const float*)d_in[1];
  const float* v = (const float*)d_in[2];
  const int*   m = (const int*)d_in[3];
  float* out = (float*)d_out;

  dim3 grid(SEQ / QBLK, 2 * HEADS);   // 8 x 32 blocks
  dim3 block(WAVES * 32);             // 256 threads = 8 wave32
  flash_attn_f16wmma<<<grid, block, 0, stream>>>(q, k, v, m, out);
}
